// NTM_88192858456612
// MI455X (gfx1250) — matrix operand order repaired and checked by reading, not compiled, not run on hardware
//
#include <hip/hip_runtime.h>

// NTM forward on MI455X (gfx1250): one workgroup per batch element; the full
// 512x64 memory matrix lives in LDS (f32 master + padded f16 shadow + f16
// transpose for WMMA). v_wmma_f32_16x16x32_f16 computes content similarity
// (Mem . k^T) and read vectors (rw @ Mem). x rows are DMA'd via the Tensor
// Data Mover (tensor_load_to_lds + s_wait_tensorcnt). T=32 recurrence is
// fully on-chip.

typedef __attribute__((ext_vector_type(16))) _Float16     v16h;
typedef __attribute__((ext_vector_type(8)))  float        v8f;
typedef __attribute__((ext_vector_type(4)))  unsigned int v4u;
typedef __attribute__((ext_vector_type(8)))  int          v8i;
typedef __attribute__((ext_vector_type(4)))  int          v4i;

#define BATCH  32
#define TSTEPS 32
#define DIN    128
#define NSLOT  512
#define MLEN   64
#define CTRL   256
#define NREAD  4
#define NHEAD  5          // 4 read heads + 1 write head
#define RHL    70         // READ_HEAD_LEN
#define INSTRD 478
#define CIN    (DIN + NREAD*MLEN)   // 384

#define MEMH_STRIDE 72    // f16 shadow row stride (halves): 144B, 16B-aligned, bank-spread
#define MEMT_STRIDE 520   // f16 transpose row stride (halves): 1040B, 16B-aligned

// ---- LDS layout: f32 region (word offsets), then f16 region (half offsets) ----
enum : int {
  OFF_MEMF  = 0,                          // 512*64 f32 memory master
  OFF_RWF   = OFF_MEMF + NSLOT*MLEN,      // 4*512 read weights f32
  OFF_WWF   = OFF_RWF  + NREAD*NSLOT,     // 512 write weights
  OFF_SIM   = OFF_WWF  + NSLOT,           // 5*512 dots; reused as gated weights
  OFF_MN    = OFF_SIM  + NHEAD*NSLOT,     // 512 row norms
  OFF_R     = OFF_MN   + NSLOT,           // 4*64 read vectors
  OFF_CTRL  = OFF_R    + NREAD*MLEN,      // 384 controller input
  OFF_H     = OFF_CTRL + CIN,             // 256 hidden
  OFF_INS   = OFF_H    + CTRL,            // 480 instr (478 used)
  OFF_KF    = OFF_INS  + 480,             // 5*64 keys f32
  OFF_EA    = OFF_KF   + NHEAD*MLEN,      // 64 erase
  OFF_AA    = OFF_EA   + MLEN,            // 64 add
  OFF_SCAL  = OFF_AA   + MLEN,            // 5*8 scalars: beta,g,s0,s1,s2,t,kn
  F32WORDS  = OFF_SCAL + NHEAD*8,         // 40264 words = 161056 B (16B aligned)
};
enum : int {  // half-word offsets within the f16 region
  HOFF_MEMH = 0,                               // [512][72]
  HOFF_MEMT = HOFF_MEMH + NSLOT*MEMH_STRIDE,   // [64][520] transpose
  HOFF_RWH  = HOFF_MEMT + MLEN*MEMT_STRIDE,    // [16][512] padded read weights
  HOFF_KH   = HOFF_RWH  + 16*NSLOT,            // [16][64] padded keys
  H16WORDS  = HOFF_KH   + 16*MLEN,
};
constexpr size_t SMEM_BYTES = size_t(F32WORDS)*4 + size_t(H16WORDS)*2; // 319776 B

__device__ __forceinline__ float waveMax(float v) {
  #pragma unroll
  for (int m = 16; m > 0; m >>= 1) v = fmaxf(v, __shfl_xor(v, m, 32));
  return v;
}
__device__ __forceinline__ float waveSum(float v) {
  #pragma unroll
  for (int m = 16; m > 0; m >>= 1) v += __shfl_xor(v, m, 32);
  return v;
}

__global__ __launch_bounds__(512, 1)
void ntm_mi455x_kernel(const float* __restrict__ x,  const float* __restrict__ Wc,
                       const float* __restrict__ bc, const float* __restrict__ hk,
                       const float* __restrict__ hb, float* __restrict__ out) {
  extern __shared__ char smem_raw[];
  float* sf   = (float*)smem_raw;
  float* MemF = sf + OFF_MEMF;
  float* rwF  = sf + OFF_RWF;
  float* wwF  = sf + OFF_WWF;
  float* simS = sf + OFF_SIM;
  float* wgB  = simS;                  // aliased: 6a rewrites sim in place
  float* MnA  = sf + OFF_MN;
  float* rF   = sf + OFF_R;
  float* cIn  = sf + OFF_CTRL;
  float* hF   = sf + OFF_H;
  float* ins  = sf + OFF_INS;
  float* kF   = sf + OFF_KF;
  float* eA   = sf + OFF_EA;
  float* aA   = sf + OFF_AA;
  float* scal = sf + OFF_SCAL;
  _Float16* h16  = (_Float16*)(sf + F32WORDS);
  _Float16* MemH = h16 + HOFF_MEMH;    // [512][MEMH_STRIDE]
  _Float16* MemT = h16 + HOFF_MEMT;    // [64][MEMT_STRIDE] (Mem transposed)
  _Float16* rwH  = h16 + HOFF_RWH;     // [16][512], rows 4..15 stay zero
  _Float16* kH   = h16 + HOFF_KH;      // [16][64], rows 5..15 stay zero

  const int tid  = threadIdx.x;
  const int lane = tid & 31;
  const int wave = tid >> 5;
  const int b    = blockIdx.x;

  // ---- init state (zero-padded operands make WMMA fragment loads branch-free)
  for (int i = tid; i < NSLOT * MLEN;        i += 512) MemF[i] = 0.f;
  for (int i = tid; i < NSLOT * MEMH_STRIDE; i += 512) MemH[i] = (_Float16)0.f;
  for (int i = tid; i < MLEN * MEMT_STRIDE;  i += 512) MemT[i] = (_Float16)0.f;
  for (int i = tid; i < 16 * NSLOT;          i += 512) rwH[i]  = (_Float16)0.f;
  for (int i = tid; i < 16 * MLEN;           i += 512) kH[i]   = (_Float16)0.f;
  for (int i = tid; i < NREAD * NSLOT;       i += 512) rwF[i]  = 0.f;
  for (int i = tid; i < NSLOT;               i += 512) wwF[i]  = 0.f;
  for (int i = tid; i < NREAD * MLEN;        i += 512) rF[i]   = 0.f;
  __syncthreads();

  float vbuf[16];

  for (int t = 0; t < TSTEPS; ++t) {
    // ---- phase 1: ctrl_in = [x_t, r];  x row via Tensor Data Mover if present
#if __has_builtin(__builtin_amdgcn_tensor_load_to_lds)
    if (wave == 0) {
      unsigned long long ga =
          (unsigned long long)(const void*)(x + (size_t)(b * TSTEPS + t) * DIN);
      v4u g0;
      g0[0] = 1u;                                   // count=1 (valid descriptor)
      g0[1] = (unsigned)(OFF_CTRL * 4);             // lds_addr (bytes)
      g0[2] = (unsigned)(ga & 0xffffffffu);         // global_addr[31:0]
      g0[3] = (unsigned)((ga >> 32) & 0x01ffffffu)  // global_addr[56:32]
              | (2u << 30);                         // type=2 ("image")
      v8i g1;
      g1[0] = 2 << 16;                              // data_size=4B
      g1[1] = DIN << 16;                            // tensor_dim0=128 (bits 79:48)
      g1[2] = 1 << 16;                              // tensor_dim1=1   (bits 111:80)
      g1[3] = DIN << 16;                            // tile_dim0=128   (bits 127:112)
      g1[4] = 1;                                    // tile_dim1=1
      g1[5] = DIN;                                  // tensor_dim0_stride=128
      g1[6] = 0; g1[7] = 0;
      v4i gz = {0, 0, 0, 0};
  #if __clang_major__ >= 23
      v8i gz8 = {0, 0, 0, 0, 0, 0, 0, 0};
      __builtin_amdgcn_tensor_load_to_lds(g0, g1, gz, gz, gz8, 0);
  #else
      __builtin_amdgcn_tensor_load_to_lds(g0, g1, gz, gz, 0);
  #endif
      __builtin_amdgcn_s_wait_tensorcnt(0);
    }
    if (tid >= DIN && tid < CIN) cIn[tid] = rF[tid - DIN];
#else
    if (tid < DIN)      cIn[tid] = x[(b * TSTEPS + t) * DIN + tid];
    else if (tid < CIN) cIn[tid] = rF[tid - DIN];
#endif
    if (t + 1 < TSTEPS && tid < DIN)
      __builtin_prefetch(&x[(b * TSTEPS + t + 1) * DIN + tid], 0, 1); // global_prefetch_b8
    __syncthreads();

    // ---- phase 2: h = tanh(ctrl_in @ Wc + bc) (f32 matvec, coalesced) ----
    if (tid < CTRL) {
      float acc = bc[tid];
      for (int i = 0; i < CIN; ++i) acc = fmaf(cIn[i], Wc[i * CTRL + tid], acc);
      hF[tid] = tanhf(acc);
    }
    __syncthreads();

    // ---- phase 3: instr = h @ heads_kernel + heads_bias ----
    if (tid < INSTRD) {
      float acc = hb[tid];
      for (int i = 0; i < CTRL; ++i) acc = fmaf(hF[i], hk[i * INSTRD + tid], acc);
      ins[tid] = acc;
    }
    __syncthreads();

    // ---- phase 4: parse heads ----
    if (tid < NHEAD * MLEN) {
      int hd = tid >> 6, m = tid & 63;
      int base = (hd < NREAD) ? hd * RHL : NREAD * RHL;
      float v = ins[base + m];
      kF[tid] = v;
      kH[hd * MLEN + m] = (_Float16)v;
    } else if (tid < NHEAD * MLEN + NHEAD) {
      int hd = tid - NHEAD * MLEN;
      int base = (hd < NREAD) ? hd * RHL : NREAD * RHL;
      scal[hd * 8 + 0] = expf(ins[base + MLEN]);                    // beta
      scal[hd * 8 + 1] = 1.f / (1.f + expf(-ins[base + MLEN + 1])); // g
      float s0 = ins[base + MLEN + 2], s1 = ins[base + MLEN + 3], s2 = ins[base + MLEN + 4];
      float mx = fmaxf(s0, fmaxf(s1, s2));
      s0 = expf(s0 - mx); s1 = expf(s1 - mx); s2 = expf(s2 - mx);
      float si = 1.f / (s0 + s1 + s2);
      scal[hd * 8 + 2] = s0 * si; scal[hd * 8 + 3] = s1 * si; scal[hd * 8 + 4] = s2 * si;
      float tv = ins[base + MLEN + 5];
      float sp = (tv > 20.f) ? tv : log1pf(expf(tv));               // softplus
      scal[hd * 8 + 5] = sp + 1.f;
    } else if (tid >= 384 && tid < 384 + 2 * MLEN) {
      int i = tid - 384;
      if (i < MLEN) eA[i] = ins[NREAD * RHL + RHL + i];
      else          aA[i - MLEN] = ins[NREAD * RHL + RHL + MLEN + (i - MLEN)];
    }
    __syncthreads();

    // ---- phase 4b: Mem row norms + key norms ----
    {
      float s = 0.f;
      #pragma unroll 8
      for (int m = 0; m < MLEN; ++m) {
        float v = MemF[tid * MLEN + ((m + lane) & 63)]; // staggered, bank-spread
        s = fmaf(v, v, s);
      }
      MnA[tid] = sqrtf(s);
    }
    if (tid < NHEAD) {
      float s = 0.f;
      for (int m = 0; m < MLEN; ++m) { float v = kF[tid * MLEN + m]; s = fmaf(v, v, s); }
      scal[tid * 8 + 6] = sqrtf(s);
    }
    __syncthreads();

    // ---- phase 5: sim[h][n] = k_h . Mem[n] via WMMA f16 (all 16 waves) ----
    {
      const int hd = lane & 15;
      const int kbB = (lane < 16) ? 0 : 16;    // B layout: contiguous K per lane half
      v16h b0, b1;
      #pragma unroll
      for (int e = 0; e < 16; ++e) {           // kH rows 5..15 are zero -> branch-free
        b0[e] = kH[hd * MLEN + kbB + e];
        b1[e] = kH[hd * MLEN + 32 + kbB + e];
      }
      for (int rt = wave * 2; rt < wave * 2 + 2; ++rt) {
        const int mrow = rt * 16 + (lane & 15);
        v16h a0, a1;
        #pragma unroll
        for (int e = 0; e < 16; ++e) {         // A layout: contiguous K per lane half
          int kb = (lane < 16) ? 0 : 16;
          int k0 = kb + e;
          a0[e] = MemH[mrow * MEMH_STRIDE + k0];
          a1[e] = MemH[mrow * MEMH_STRIDE + 32 + k0];
        }
        v8f c = {};
        c = __builtin_amdgcn_wmma_f32_16x16x32_f16(false, a0, false, b0, (short)0, c, false, false);
        c = __builtin_amdgcn_wmma_f32_16x16x32_f16(false, a1, false, b1, (short)0, c, false, false);
        if (hd < NHEAD) {
          #pragma unroll
          for (int v = 0; v < 8; ++v) {
            int n = rt * 16 + v + ((lane < 16) ? 0 : 8);
            simS[hd * NSLOT + n] = c[v];
          }
        }
      }
    }
    __syncthreads();

    // ---- phase 6a: cosine sim, softmax(beta*sim), gate (wave w -> head w) ----
    if (wave < NHEAD) {
      const int hd = wave;
      const float bet = scal[hd * 8 + 0], g = scal[hd * 8 + 1], kn = scal[hd * 8 + 6];
      const float* wprev = (hd < NREAD) ? (rwF + hd * NSLOT) : wwF;
      float mx = -1e30f;
      #pragma unroll
      for (int j = 0; j < 16; ++j) {
        int n = j * 32 + lane;
        float z = bet * (simS[hd * NSLOT + n] / (kn * MnA[n] + 1e-8f));
        vbuf[j] = z; mx = fmaxf(mx, z);
      }
      mx = waveMax(mx);
      float sum = 0.f;
      #pragma unroll
      for (int j = 0; j < 16; ++j) { vbuf[j] = expf(vbuf[j] - mx); sum += vbuf[j]; }
      sum = waveSum(sum);
      const float inv = 1.f / sum;
      #pragma unroll
      for (int j = 0; j < 16; ++j) {           // in-place: each slot read then written by same lane
        int n = j * 32 + lane;
        wgB[hd * NSLOT + n] = g * (vbuf[j] * inv) + (1.f - g) * wprev[n];
      }
    }
    __syncthreads();

    // ---- phase 6b: circular shift, sharpen (^t), renormalize ----
    if (wave < NHEAD) {
      const int hd = wave;
      const float s0 = scal[hd * 8 + 2], s1 = scal[hd * 8 + 3], s2 = scal[hd * 8 + 4];
      const float tp = scal[hd * 8 + 5];
      float* wdst = (hd < NREAD) ? (rwF + hd * NSLOT) : wwF;
      const float* wg = wgB + hd * NSLOT;
      float sum = 0.f;
      #pragma unroll
      for (int j = 0; j < 16; ++j) {
        int n = j * 32 + lane;
        float w = s0 * wg[(n + 1) & 511] + s1 * wg[n] + s2 * wg[(n - 1) & 511];
        float wt = powf(w, tp);
        vbuf[j] = wt; sum += wt;
      }
      sum = waveSum(sum);
      const float inv = 1.f / (sum + 1e-8f);
      #pragma unroll
      for (int j = 0; j < 16; ++j) {
        int n = j * 32 + lane;
        float wv = vbuf[j] * inv;
        wdst[n] = wv;
        if (hd < NREAD) rwH[hd * NSLOT + n] = (_Float16)wv;  // f16 shadow for WMMA
      }
    }
    __syncthreads();

    // ---- phase 7: Mem = Mem*(1 - ww*e) + ww*a  (f32 master + f16 shadow + transpose)
    #pragma unroll 4
    for (int j = 0; j < (NSLOT * MLEN) / 512; ++j) {
      int p = j * 512 + tid;
      int n = p >> 6, m = p & 63;
      float w = wwF[n];
      float v = MemF[p] * (1.f - w * eA[m]) + w * aA[m];
      MemF[p] = v;
      _Float16 vh = (_Float16)v;
      MemH[n * MEMH_STRIDE + m] = vh;
      MemT[m * MEMT_STRIDE + n] = vh;
    }
    __syncthreads();

    // ---- phase 8: r[h] = sum_n rw[h][n]*Mem[n] via WMMA (waves 0..3) ----
    if (wave < 4) {
      const int mrow = lane & 15;
      const int ncol = wave * 16 + (lane & 15);
      v8f c = {};
      for (int kk = 0; kk < 16; ++kk) {
        const int kbA = kk * 32 + ((lane < 16) ? 0 : 16);
        const int kbB = kk * 32 + ((lane < 16) ? 0 : 16);
        v16h a, bb;
        #pragma unroll
        for (int e = 0; e < 16; ++e) {
          a[e]  = rwH[mrow * NSLOT + kbA + e];                     // rows 4..15 zero
          bb[e] = MemT[ncol * MEMT_STRIDE + kbB + e];              // contiguous
        }
        c = __builtin_amdgcn_wmma_f32_16x16x32_f16(false, a, false, bb, (short)0, c, false, false);
      }
      if (lane < 16) {                 // rows 0..3 of D are the 4 read heads
        #pragma unroll
        for (int v = 0; v < NREAD; ++v) rF[v * MLEN + ncol] = c[v];
      }
    }
    __syncthreads();
  }

  // output: h at final timestep
  if (tid < CTRL) out[b * CTRL + tid] = hF[tid];
}

extern "C" void kernel_launch(void* const* d_in, const int* in_sizes, int n_in,
                              void* d_out, int out_size, void* d_ws, size_t ws_size,
                              hipStream_t stream) {
  (void)in_sizes; (void)n_in; (void)d_ws; (void)ws_size; (void)out_size;
  const float* x  = (const float*)d_in[0];
  const float* Wc = (const float*)d_in[1];
  const float* bc = (const float*)d_in[2];
  const float* hk = (const float*)d_in[3];
  const float* hb = (const float*)d_in[4];
  float* out = (float*)d_out;

  // Allow >64KB dynamic LDS (CDNA5 WGP has 320KB). Deterministic, capture-safe.
  (void)hipFuncSetAttribute(reinterpret_cast<const void*>(ntm_mi455x_kernel),
                            hipFuncAttributeMaxDynamicSharedMemorySize,
                            (int)SMEM_BYTES);
  ntm_mi455x_kernel<<<BATCH, 512, SMEM_BYTES, stream>>>(x, Wc, bc, hk, hb, out);
}
